// Attention_NMT_80547816669399
// MI455X (gfx1250) — matrix operand order repaired and checked by reading, not compiled, not run on hardware
//
#include <hip/hip_runtime.h>
#include <hip/hip_bf16.h>

// ---------------------------------------------------------------------------
// Types for CDNA5 WMMA (wave32, v_wmma_f32_16x16x32_f16)
// ---------------------------------------------------------------------------
typedef __attribute__((ext_vector_type(16))) _Float16 v16h;
typedef __attribute__((ext_vector_type(8)))  _Float16 v8h;
typedef __attribute__((ext_vector_type(8)))  float    v8f;

#define BM 64
#define BN 128
#define BK 32
#define ALD 40   // LDS row stride (halves) for A  : 80 B, 16B aligned
#define WLD 40   // LDS row stride (halves) for W^T: 80 B, 16B aligned

// Generic GEMM: C[M,N] = A[M,K] @ W[K,N] (+bias) (+=C if flags&1) (relu if flags&2)
// fp32 operands converted to f16 in LDS; fp32 WMMA accumulation.
// Requires: M % 64 == 0, N % 128 == 0, K % 32 == 0 (true for every GEMM below).
//
// LDW is a template parameter so the 8 column-wise W loads per chunk lower to
// independent immediate-offset global_load_b32 (claused), not a v_add_u64 /
// s_wait_xcnt pointer chain. Max offset 7*32000*4 B fits the signed 24-bit
// instruction offset.
//
// LDS layouts make every fragment load a ds_load_b128:
//   As  [64][ALD]  K-major  -> A frag = 2 x b128 per lane (ISA 7.12.2 layout)
//   WsT [128][WLD] N-major  -> B frag = 2 x b128 per lane per col-tile
template<int LDW>
__global__ __launch_bounds__(256) void gemm_f16_wmma(
    const float* __restrict__ A, int lda,
    const float* __restrict__ W,
    float* __restrict__ C, int ldc,
    const float* __restrict__ bias,
    int K, int flags)
{
    __shared__ _Float16 As[BM][ALD]  __attribute__((aligned(16)));
    __shared__ _Float16 WsT[BN][WLD] __attribute__((aligned(16)));

    const int tid  = threadIdx.x;
    const int lane = tid & 31;
    const int wave = tid >> 5;
    const int bm   = blockIdx.y * BM;
    const int bn   = blockIdx.x * BN;

    // 8 waves = 4 row-tiles x 2 col-halves; each wave: 1 row-tile x 4 col-tiles
    const int wr  = (wave & 3) * 16;
    const int wcb = (wave >> 2) * 64;

    const int half = lane >> 4;     // 0: lanes 0-15, 1: lanes 16-31
    const int l15  = lane & 15;

    // A staging map: thread -> (row, 8-col chunk) of the 64x32 tile
    const int s_ar = tid >> 2;          // 0..63
    const int s_ac = (tid & 3) * 8;     // 0,8,16,24
    // W staging map: thread -> column n (two kq chunks of 8)
    const int s_wn  = tid & 127;        // 0..127
    const int s_wk0 = (tid >> 7) * 8;   // 0 or 8  (second chunk = +16)

    v8f acc0 = {}, acc1 = {}, acc2 = {}, acc3 = {};

    for (int k0 = 0; k0 < K; k0 += BK) {
        // ---- stage A (64x32): 2x float4 global load -> 1x b128 LDS store ----
        {
            const float* ap = A + (long)(bm + s_ar) * lda + k0 + s_ac;
            float4 v0 = reinterpret_cast<const float4*>(ap)[0];
            float4 v1 = reinterpret_cast<const float4*>(ap)[1];
            v8h p;
            p[0] = (_Float16)v0.x; p[1] = (_Float16)v0.y;
            p[2] = (_Float16)v0.z; p[3] = (_Float16)v0.w;
            p[4] = (_Float16)v1.x; p[5] = (_Float16)v1.y;
            p[6] = (_Float16)v1.z; p[7] = (_Float16)v1.w;
            *reinterpret_cast<v8h*>(&As[s_ar][s_ac]) = p;
        }
        // ---- stage W (32x128) transposed: per column, 8 coalesced b32 loads
        //      (constant offsets i*LDW) -> 1x b128 LDS store; 2 chunks/thread ----
        #pragma unroll
        for (int u = 0; u < 2; ++u) {
            const int kq = s_wk0 + u * 16;
            const float* wp = W + (long)(k0 + kq) * LDW + bn + s_wn;
            v8h p;
            #pragma unroll
            for (int i = 0; i < 8; ++i)
                p[i] = (_Float16)wp[i * LDW];
            *reinterpret_cast<v8h*>(&WsT[s_wn][kq]) = p;
        }
        // prefetch next K tile while this one is consumed
        if (k0 + BK < K) {
            __builtin_prefetch(A + (long)(bm + s_ar) * lda + k0 + BK + s_ac, 0, 1);
            __builtin_prefetch(W + (long)(k0 + BK + s_wk0) * LDW + bn + s_wn, 0, 1);
        }
        __syncthreads();

        // ---- fragments (all b128 LDS loads) ----
        const int am = wr + l15;
        const int kb = half * 8;    // lanes 0-15: K 0..7/16..23; 16-31: K 8..15/24..31
        v8h a_lo = *reinterpret_cast<const v8h*>(&As[am][kb]);
        v8h a_hi = *reinterpret_cast<const v8h*>(&As[am][16 + kb]);
        v16h af = __builtin_shufflevector(a_lo, a_hi,
                 0,1,2,3,4,5,6,7,8,9,10,11,12,13,14,15);

        const int kbb = half * 8;   // lanes 0-15: K 0..7/16..23; 16-31: K 8..15/24..31
        #pragma unroll
        for (int j = 0; j < 4; ++j) {
            const int col = wcb + j * 16 + l15;
            v8h b_lo = *reinterpret_cast<const v8h*>(&WsT[col][kbb]);
            v8h b_hi = *reinterpret_cast<const v8h*>(&WsT[col][16 + kbb]);
            v16h bf = __builtin_shufflevector(b_lo, b_hi,
                     0,1,2,3,4,5,6,7,8,9,10,11,12,13,14,15);
            if (j == 0) acc0 = __builtin_amdgcn_wmma_f32_16x16x32_f16(false, af, false, bf, (short)0, acc0, false, false);
            if (j == 1) acc1 = __builtin_amdgcn_wmma_f32_16x16x32_f16(false, af, false, bf, (short)0, acc1, false, false);
            if (j == 2) acc2 = __builtin_amdgcn_wmma_f32_16x16x32_f16(false, af, false, bf, (short)0, acc2, false, false);
            if (j == 3) acc3 = __builtin_amdgcn_wmma_f32_16x16x32_f16(false, af, false, bf, (short)0, acc3, false, false);
        }
        __syncthreads();
    }

    // ---- epilogue: C/D layout = VGPR r holds rows r (lanes 0-15) / r+8 (16-31)
    #pragma unroll
    for (int j = 0; j < 4; ++j) {
        const v8f acc = (j == 0) ? acc0 : (j == 1) ? acc1 : (j == 2) ? acc2 : acc3;
        const int col = bn + wcb + j * 16 + l15;
        const float bj = bias ? bias[col] : 0.f;
        #pragma unroll
        for (int r = 0; r < 8; ++r) {
            const int row = bm + wr + r + 8 * half;
            float v = acc[r] + bj;
            float* p = C + (long)row * ldc + col;
            if (flags & 1) v += *p;
            if (flags & 2) v = fmaxf(v, 0.f);
            *p = v;
        }
    }
}

// ---------------------------------------------------------------------------
// Elementwise / support kernels
// ---------------------------------------------------------------------------
__global__ void fill_kernel(float* p, long n, float v)
{
    long i = (long)blockIdx.x * blockDim.x + threadIdx.x;
    if (i < n) p[i] = v;
}

// out[r * out_stride + e] = table[idx[r] * E + e]
__global__ void gather_rows(const float* __restrict__ table,
                            const int* __restrict__ idx,
                            long n_rows, int E,
                            float* __restrict__ out, long out_stride)
{
    long i = (long)blockIdx.x * blockDim.x + threadIdx.x;
    if (i >= n_rows * E) return;
    long r = i / E;
    int  e = (int)(i - r * E);
    out[r * out_stride + e] = table[(long)idx[r] * E + e];
}

// ci[b, t, 2560 + j] = graph_embs[b, j]   (B=64, T=64, G=256, ldc=3328)
__global__ void broadcast_graph(const float* __restrict__ g, float* __restrict__ ci)
{
    long i = (long)blockIdx.x * blockDim.x + threadIdx.x;
    if (i >= (long)64 * 64 * 256) return;
    int b = (int)(i >> 14);
    int rem = (int)(i & 16383);
    int t = rem >> 8;
    int j = rem & 255;
    ci[((long)b * 64 + t) * 3328 + 2560 + j] = g[b * 256 + j];
}

__device__ __forceinline__ float sigmoidf_(float x) { return 1.f / (1.f + __expf(-x)); }

// gates [64, 2048] -> h,c [64,512]; also mirror h into out2[b*stride + j]
__global__ void lstm_cell_kernel(const float* __restrict__ gates,
                                 float* __restrict__ h, float* __restrict__ c,
                                 float* __restrict__ out2, long out2_stride)
{
    long i = (long)blockIdx.x * blockDim.x + threadIdx.x;
    if (i >= (long)64 * 512) return;
    int b = (int)(i >> 9);
    int j = (int)(i & 511);
    const float* g = gates + (long)b * 2048;
    float gi = g[j];
    float gf = g[512 + j];
    float gg = g[1024 + j];
    float go = g[1536 + j];
    float cn = sigmoidf_(gf) * c[i] + sigmoidf_(gi) * tanhf(gg);
    float hn = sigmoidf_(go) * tanhf(cn);
    c[i] = cn;
    h[i] = hn;
    if (out2) out2[(long)b * out2_stride + j] = hn;
}

// Fused attention scoring + softmax + context; one batch row per block.
// scores[s] = b2 + sum_k relu(enc_part[b,s,k] + hpart[b,k]) * W2[k]
// a[b,:]    = sum_s softmax(scores)[s] * enc_out[b,s,:]
__global__ __launch_bounds__(256) void attention_kernel(
    const float* __restrict__ enc_part,   // [64,64,1536] (b1 already added)
    const float* __restrict__ hpart,      // [64,1536]
    const float* __restrict__ enc_out,    // [64,64,1024]
    const float* __restrict__ W2,         // [1536]
    const float* __restrict__ b2p,        // [1]
    float* __restrict__ aout)             // [64,1024]
{
    const int b   = blockIdx.x;
    const int tid = threadIdx.x;
    __shared__ float hp[1536];
    __shared__ float red[256];
    __shared__ float sc[64];

    for (int k = tid; k < 1536; k += 256) hp[k] = hpart[(long)b * 1536 + k];
    __syncthreads();

    const int s = tid >> 2;   // 0..63
    const int q = tid & 3;
    const float* ep = enc_part + ((long)b * 64 + s) * 1536;
    float part = 0.f;
    for (int k = q; k < 1536; k += 4) {
        float v = fmaxf(ep[k] + hp[k], 0.f);
        part += v * W2[k];
    }
    red[tid] = part;
    __syncthreads();
    if (q == 0) sc[s] = red[tid] + red[tid + 1] + red[tid + 2] + red[tid + 3] + b2p[0];
    __syncthreads();

    if (tid == 0) {
        float m = sc[0];
        for (int i = 1; i < 64; ++i) m = fmaxf(m, sc[i]);
        float sum = 0.f;
        for (int i = 0; i < 64; ++i) { float e = __expf(sc[i] - m); sc[i] = e; sum += e; }
        float inv = 1.f / sum;
        for (int i = 0; i < 64; ++i) sc[i] *= inv;
    }
    __syncthreads();

    for (int j = tid; j < 1024; j += 256) {
        const float* eo = enc_out + (long)b * 64 * 1024 + j;
        float acc = 0.f;
        #pragma unroll 8
        for (int s2 = 0; s2 < 64; ++s2) acc += sc[s2] * eo[s2 * 1024];
        aout[(long)b * 1024 + j] = acc;
    }
}

// Build decoder input x[b,:] = [tgt_e(b,t), a(b,:), ar(b,:)] and scatter a,ar
// into the classifier input ci at columns [512,1536) and [1536,2560).
__global__ void assemble_dec_input(float* __restrict__ ci, int t,
                                   const float* __restrict__ a,
                                   const float* __restrict__ ar,
                                   float* __restrict__ x)
{
    long i = (long)blockIdx.x * blockDim.x + threadIdx.x;
    if (i >= (long)64 * 2560) return;
    int b = (int)(i / 2560);
    int k = (int)(i - (long)b * 2560);
    float* crow = ci + ((long)b * 64 + t) * 3328;
    float v;
    if (k < 512) {
        v = crow[k];                       // tgt embedding already gathered into ci
    } else if (k < 1536) {
        v = a[(long)b * 1024 + (k - 512)];
        crow[k] = v;
    } else {
        v = ar[(long)b * 1024 + (k - 1536)];
        crow[k] = v;
    }
    x[i] = v;
}

// ---------------------------------------------------------------------------
// Host orchestration
// ---------------------------------------------------------------------------
extern "C" void kernel_launch(void* const* d_in, const int* in_sizes, int n_in,
                              void* d_out, int out_size, void* d_ws, size_t ws_size,
                              hipStream_t stream)
{
    (void)in_sizes; (void)n_in; (void)out_size; (void)ws_size;

    const int*   source_data = (const int*)  d_in[0];
    const int*   target_data = (const int*)  d_in[1];
    const int*   rationales  = (const int*)  d_in[2];
    const float* graph_embs  = (const float*)d_in[3];
    const float* src_emb     = (const float*)d_in[4];
    const float* tgt_emb     = (const float*)d_in[5];
    const float* enc_Wih_f   = (const float*)d_in[6];
    const float* enc_Whh_f   = (const float*)d_in[7];
    const float* enc_b_f     = (const float*)d_in[8];
    const float* enc_Wih_b   = (const float*)d_in[9];
    const float* enc_Whh_b   = (const float*)d_in[10];
    const float* enc_b_b     = (const float*)d_in[11];
    const float* dec_Wih     = (const float*)d_in[12];
    const float* dec_Whh     = (const float*)d_in[13];
    const float* dec_b       = (const float*)d_in[14];
    const float* att_W1      = (const float*)d_in[15];
    const float* att_b1      = (const float*)d_in[16];
    const float* att_W2      = (const float*)d_in[17];
    const float* att_b2      = (const float*)d_in[18];
    const float* cls_Wg      = (const float*)d_in[19];
    const float* cls_bg      = (const float*)d_in[20];
    const float* cls_W2      = (const float*)d_in[21];
    const float* cls_b2      = (const float*)d_in[22];
    float* out = (float*)d_out;

    // ---- workspace layout (fp32 elements) ----
    size_t off = 0;
    float* base = (float*)d_ws;
    auto alloc = [&](size_t n) { float* p = base + off; off += n; return p; };
    float* src_e      = alloc((size_t)64 * 64 * 512);
    float* rat_e      = alloc((size_t)64 * 64 * 512);
    float* enc_out    = alloc((size_t)64 * 64 * 1024);
    float* enc_out_r  = alloc((size_t)64 * 64 * 1024);
    float* enc_part   = alloc((size_t)64 * 64 * 1536);
    float* enc_part_r = alloc((size_t)64 * 64 * 1536);
    float* ci         = alloc((size_t)64 * 64 * 3328);
    float* cls_hidden = alloc((size_t)64 * 64 * 1024);
    float* gates      = alloc((size_t)64 * 2048);
    float* xdec       = alloc((size_t)64 * 2560);
    float* abuf       = alloc((size_t)64 * 1024);
    float* arbuf      = alloc((size_t)64 * 1024);
    float* hA         = alloc((size_t)64 * 512);
    float* cA         = alloc((size_t)64 * 512);
    float* dec_h      = alloc((size_t)64 * 512);
    float* dec_c      = alloc((size_t)64 * 512);
    float* hpart      = alloc((size_t)64 * 1536);

    // LDW-specialized dispatch (values used by this model only)
    auto gemm = [&](const float* A, int lda, const float* W, int ldw,
                    float* C, int ldc, const float* bias,
                    int M, int N, int K, int flags) {
        dim3 grid(N / BN, M / BM);
        switch (ldw) {
        case 1024:
            gemm_f16_wmma<1024><<<grid, 256, 0, stream>>>(A, lda, W, C, ldc, bias, K, flags);
            break;
        case 1536:
            gemm_f16_wmma<1536><<<grid, 256, 0, stream>>>(A, lda, W, C, ldc, bias, K, flags);
            break;
        case 2048:
            gemm_f16_wmma<2048><<<grid, 256, 0, stream>>>(A, lda, W, C, ldc, bias, K, flags);
            break;
        case 32000:
            gemm_f16_wmma<32000><<<grid, 256, 0, stream>>>(A, lda, W, C, ldc, bias, K, flags);
            break;
        }
    };
    auto fill = [&](float* p, long n, float v) {
        fill_kernel<<<(int)((n + 255) / 256), 256, 0, stream>>>(p, n, v);
    };

    // ---- embeddings ----
    gather_rows<<<(64 * 64 * 512) / 256, 256, 0, stream>>>(src_emb, source_data,
                                                           64 * 64, 512, src_e, 512);
    gather_rows<<<(64 * 64 * 512) / 256, 256, 0, stream>>>(src_emb, rationales,
                                                           64 * 64, 512, rat_e, 512);
    // classifier input: zero, then tgt embeddings (cols 0..511) + graph (2560..2815)
    fill(ci, (long)64 * 64 * 3328, 0.f);
    gather_rows<<<(64 * 64 * 512) / 256, 256, 0, stream>>>(tgt_emb, target_data,
                                                           64 * 64, 512, ci, 3328);
    broadcast_graph<<<(64 * 64 * 256) / 256, 256, 0, stream>>>(graph_embs, ci);

    // ---- bidirectional encoder LSTMs ----
    auto run_dir = [&](const float* xbase, bool rev,
                       const float* Wih, const float* Whh, const float* bias,
                       float* hb, float* cb, float* ybase /* enc_out + col_off */) {
        fill(hb, 64 * 512, 0.f);
        fill(cb, 64 * 512, 0.f);
        for (int s = 0; s < 64; ++s) {
            int sp = rev ? 63 - s : s;
            gemm(xbase + (long)sp * 512, 64 * 512, Wih, 2048, gates, 2048,
                 bias, 64, 2048, 512, 0);
            gemm(hb, 512, Whh, 2048, gates, 2048, nullptr, 64, 2048, 512, 1);
            lstm_cell_kernel<<<128, 256, 0, stream>>>(gates, hb, cb,
                                                      ybase + (long)sp * 1024,
                                                      (long)64 * 1024);
        }
    };
    // src fwd run keeps its final (h,c) in dec_h/dec_c -> decoder init state
    run_dir(src_e, false, enc_Wih_f, enc_Whh_f, enc_b_f, dec_h, dec_c, enc_out);
    run_dir(src_e, true,  enc_Wih_b, enc_Whh_b, enc_b_b, hA, cA, enc_out + 512);
    run_dir(rat_e, false, enc_Wih_f, enc_Whh_f, enc_b_f, hA, cA, enc_out_r);
    run_dir(rat_e, true,  enc_Wih_b, enc_Whh_b, enc_b_b, hA, cA, enc_out_r + 512);

    // ---- hoisted attention precompute: enc_out @ W1[:2H] + b1 (loop invariant) ----
    gemm(enc_out,   1024, att_W1, 1536, enc_part,   1536, att_b1, 4096, 1536, 1024, 0);
    gemm(enc_out_r, 1024, att_W1, 1536, enc_part_r, 1536, att_b1, 4096, 1536, 1024, 0);
    const float* att_W1_h = att_W1 + (size_t)1024 * 1536;   // rows [2H,3H)

    // ---- decoder: 32 steps ----
    for (int t = 0; t < 32; ++t) {
        gemm(dec_h, 512, att_W1_h, 1536, hpart, 1536, nullptr, 64, 1536, 512, 0);
        attention_kernel<<<64, 256, 0, stream>>>(enc_part,   hpart, enc_out,
                                                 att_W2, att_b2, abuf);
        attention_kernel<<<64, 256, 0, stream>>>(enc_part_r, hpart, enc_out_r,
                                                 att_W2, att_b2, arbuf);
        assemble_dec_input<<<(64 * 2560) / 256, 256, 0, stream>>>(ci, t, abuf, arbuf, xdec);
        gemm(xdec, 2560, dec_Wih, 2048, gates, 2048, dec_b, 64, 2048, 2560, 0);
        gemm(dec_h, 512, dec_Whh, 2048, gates, 2048, nullptr, 64, 2048, 512, 1);
        lstm_cell_kernel<<<128, 256, 0, stream>>>(gates, dec_h, dec_c,
                                                  ci + (long)t * 3328 + 2816,
                                                  (long)64 * 3328);
    }

    // ---- classifier: the FLOP-dominant GEMMs ----
    gemm(ci, 3328, cls_Wg, 1024, cls_hidden, 1024, cls_bg, 4096, 1024, 3328, 2 /*relu*/);
    gemm(cls_hidden, 1024, cls_W2, 32000, out, 32000, cls_b2, 4096, 32000, 1024, 0);
}
